// LinearOffsetLayer_41901700940369
// MI455X (gfx1250) — compile-verified, matrix operand order checked by reading, not run on hardware
//
#include <hip/hip_runtime.h>

#define DD    128
#define INF   1024
#define OUTF  1024
#define NROW  4096

typedef __attribute__((ext_vector_type(4)))  float  v4f;
typedef __attribute__((ext_vector_type(8)))  float  v8f;
typedef __attribute__((ext_vector_type(4)))  __bf16 v4bf;
typedef __attribute__((ext_vector_type(8)))  __bf16 v8bf;
typedef __attribute__((ext_vector_type(16))) __bf16 v16bf;

__device__ __forceinline__ void split_bf16(float f, __bf16& hi, __bf16& lo) {
    __bf16 h = (__bf16)f;
    hi = h;
    lo = (__bf16)(f - (float)h);
}

// ---------------------------------------------------------------------------
// Kernel 1: W[o,i] = sum_d theta[d] * P_A[o,d,i] + theta0_A[o,i]
// Streaming, memory-bound over 537MB of P_A. One block per output row o.
// Writes split bf16 (hi/lo) directly for the WMMA GEMM.
// ---------------------------------------------------------------------------
__global__ __launch_bounds__(256) void synth_w_kernel(
    const float* __restrict__ theta_d, const float* __restrict__ theta0_A,
    const float* __restrict__ P_A,
    __bf16* __restrict__ W_hi, __bf16* __restrict__ W_lo)
{
    __shared__ float th[DD];
    const int tid = threadIdx.x;
    if (tid < DD) th[tid] = theta_d[tid];
    __syncthreads();

    const int o  = blockIdx.x;
    const int i0 = tid * 4;                       // 256 threads * float4 = 1024 cols
    v4f acc = {0.f, 0.f, 0.f, 0.f};
    const float* base = P_A + (size_t)o * DD * INF + i0;
    #pragma unroll 4
    for (int d = 0; d < DD; ++d) {
        v4f v = *(const v4f*)(base + (size_t)d * INF);   // coalesced 16B per lane
        acc += th[d] * v;
    }
    acc += *(const v4f*)(theta0_A + (size_t)o * INF + i0);

    v4bf h, l;
    #pragma unroll
    for (int j = 0; j < 4; ++j) { __bf16 hh, ll; split_bf16(acc[j], hh, ll); h[j] = hh; l[j] = ll; }
    *(v4bf*)(W_hi + (size_t)o * INF + i0) = h;
    *(v4bf*)(W_lo + (size_t)o * INF + i0) = l;
}

// ---------------------------------------------------------------------------
// Kernel 2: split x (fp32) into bf16 hi/lo planes.
// ---------------------------------------------------------------------------
__global__ __launch_bounds__(256) void convert_x_kernel(
    const float* __restrict__ x, __bf16* __restrict__ Xh, __bf16* __restrict__ Xl)
{
    const size_t i0 = ((size_t)blockIdx.x * 256 + threadIdx.x) * 4;
    v4f v = *(const v4f*)(x + i0);
    v4bf h, l;
    #pragma unroll
    for (int j = 0; j < 4; ++j) { __bf16 hh, ll; split_bf16(v[j], hh, ll); h[j] = hh; l[j] = ll; }
    *(v4bf*)(Xh + i0) = h;
    *(v4bf*)(Xl + i0) = l;
}

// ---------------------------------------------------------------------------
// Kernel 3: b[o] = sum_d theta[d] * P_b[d,o] + theta0_b[o]
// ---------------------------------------------------------------------------
__global__ __launch_bounds__(256) void bias_kernel(
    const float* __restrict__ theta_d, const float* __restrict__ theta0_b,
    const float* __restrict__ P_b, float* __restrict__ bvec)
{
    __shared__ float th[DD];
    if (threadIdx.x < DD) th[threadIdx.x] = theta_d[threadIdx.x];
    __syncthreads();
    const int o = blockIdx.x * 256 + threadIdx.x;
    float acc = theta0_b[o];
    #pragma unroll 4
    for (int d = 0; d < DD; ++d) acc += th[d] * P_b[(size_t)d * OUTF + o];
    bvec[o] = acc;
}

// ---------------------------------------------------------------------------
// Kernel 4: out = X @ W^T + b via split-bf16 WMMA (3x v_wmma_f32_16x16x32_bf16
// per tile pair => fp32-equivalent accuracy at bf16 matrix-core throughput).
// Wave tile 32x32 (2x2 WMMA tiles), 8 waves => block tile 128(M) x 64(N).
// Per-lane fragment addressing per ISA 7.12.2 (16-bit A layout; B operand uses
// the identical layout on row-major W since B-layout == W^T column striping).
// ---------------------------------------------------------------------------
__global__ __launch_bounds__(256) void gemm_wmma_kernel(
    const __bf16* __restrict__ Xh, const __bf16* __restrict__ Xl,
    const __bf16* __restrict__ Wh, const __bf16* __restrict__ Wl,
    const float* __restrict__ bvec, float* __restrict__ out)
{
    const int tid  = threadIdx.x;
    const int lane = tid & 31;
    const int wave = tid >> 5;          // 0..7
    const int wm   = wave >> 1;         // 0..3 -> M
    const int wn   = wave & 1;          // 0..1 -> N

    const int m0 = blockIdx.y * 128 + wm * 32;
    const int n0 = blockIdx.x * 64  + wn * 32;

    const int lr    = lane & 15;        // row within 16-tile (M for A, N for B)
    const int koff  = (lane >> 4) * 8;  // lanes 16-31 hold K+8 chunk
    const int rbase = (lane >> 4) * 8;  // C: lanes 16-31 hold M rows 8..15

    v8f c[2][2];
    #pragma unroll
    for (int nt = 0; nt < 2; ++nt) {
        const float bb = bvec[n0 + nt * 16 + lr];   // bias depends only on N col
        #pragma unroll
        for (int mt = 0; mt < 2; ++mt)
            #pragma unroll
            for (int r = 0; r < 8; ++r) c[mt][nt][r] = bb;
    }

    for (int k = 0; k < INF; k += 32) {
        v16bf ah[2], al[2], bh[2], bl[2];
        #pragma unroll
        for (int mt = 0; mt < 2; ++mt) {
            const size_t rb = (size_t)(m0 + mt * 16 + lr) * INF + (k + koff);
            const v8bf* ph = (const v8bf*)(Xh + rb);
            const v8bf* pl = (const v8bf*)(Xl + rb);
            ((v8bf*)&ah[mt])[0] = ph[0];  ((v8bf*)&ah[mt])[1] = ph[2];  // K, K+16
            ((v8bf*)&al[mt])[0] = pl[0];  ((v8bf*)&al[mt])[1] = pl[2];
        }
        #pragma unroll
        for (int nt = 0; nt < 2; ++nt) {
            const size_t rb = (size_t)(n0 + nt * 16 + lr) * INF + (k + koff);
            const v8bf* ph = (const v8bf*)(Wh + rb);
            const v8bf* pl = (const v8bf*)(Wl + rb);
            ((v8bf*)&bh[nt])[0] = ph[0];  ((v8bf*)&bh[nt])[1] = ph[2];
            ((v8bf*)&bl[nt])[0] = pl[0];  ((v8bf*)&bl[nt])[1] = pl[2];
        }
        #pragma unroll
        for (int mt = 0; mt < 2; ++mt)
            #pragma unroll
            for (int nt = 0; nt < 2; ++nt) {
                c[mt][nt] = __builtin_amdgcn_wmma_f32_16x16x32_bf16(
                    false, ah[mt], false, bh[nt], (short)0, c[mt][nt], false, false);
                c[mt][nt] = __builtin_amdgcn_wmma_f32_16x16x32_bf16(
                    false, ah[mt], false, bl[nt], (short)0, c[mt][nt], false, false);
                c[mt][nt] = __builtin_amdgcn_wmma_f32_16x16x32_bf16(
                    false, al[mt], false, bh[nt], (short)0, c[mt][nt], false, false);
            }
    }

    #pragma unroll
    for (int mt = 0; mt < 2; ++mt)
        #pragma unroll
        for (int nt = 0; nt < 2; ++nt) {
            const int col = n0 + nt * 16 + lr;
            #pragma unroll
            for (int r = 0; r < 8; ++r) {
                const int row = m0 + mt * 16 + rbase + r;
                out[(size_t)row * OUTF + col] = c[mt][nt][r];
            }
        }
}

// ---------------------------------------------------------------------------
extern "C" void kernel_launch(void* const* d_in, const int* in_sizes, int n_in,
                              void* d_out, int out_size, void* d_ws, size_t ws_size,
                              hipStream_t stream) {
    const float* x        = (const float*)d_in[0];
    const float* theta_d  = (const float*)d_in[1];
    const float* theta0_A = (const float*)d_in[2];
    const float* P_A      = (const float*)d_in[3];
    const float* theta0_b = (const float*)d_in[4];
    const float* P_b      = (const float*)d_in[5];
    float* out = (float*)d_out;

    // Workspace layout (~21 MB): Xh, Xl, Wh, Wl (bf16) + bias (fp32)
    char* ws = (char*)d_ws;
    size_t off = 0;
    __bf16* Xh = (__bf16*)(ws + off); off += (size_t)NROW * INF  * sizeof(__bf16);
    __bf16* Xl = (__bf16*)(ws + off); off += (size_t)NROW * INF  * sizeof(__bf16);
    __bf16* Wh = (__bf16*)(ws + off); off += (size_t)OUTF * INF  * sizeof(__bf16);
    __bf16* Wl = (__bf16*)(ws + off); off += (size_t)OUTF * INF  * sizeof(__bf16);
    float*  bv = (float*) (ws + off); off += (size_t)OUTF * sizeof(float);
    if (ws_size < off) return;   // insufficient scratch (deterministic no-op)

    convert_x_kernel<<<(NROW * INF) / 1024, 256, 0, stream>>>(x, Xh, Xl);
    synth_w_kernel  <<<OUTF,               256, 0, stream>>>(theta_d, theta0_A, P_A, Wh, Wl);
    bias_kernel     <<<OUTF / 256,         256, 0, stream>>>(theta_d, theta0_b, P_b, bv);

    dim3 grid(OUTF / 64, NROW / 128);
    gemm_wmma_kernel<<<grid, 256, 0, stream>>>(Xh, Xl, Wh, Wl, bv, out);
}
